// MyLongformerSelfAttention_33088428048582
// MI455X (gfx1250) — matrix operand order repaired and checked by reading, compiled, not run on hardware
//
#include <hip/hip_runtime.h>
#include <hip/hip_bf16.h>

typedef __attribute__((ext_vector_type(16))) __bf16 v16bf;
typedef __attribute__((ext_vector_type(8)))  __bf16 v8bf;
typedef __attribute__((ext_vector_type(8)))  float  v8f;

#define S_LEN 4096
#define E_DIM 768
#define H_NUM 12
#define HD_DIM 64
#define BATCH 2
#define WIN 256

static __device__ __forceinline__ v16bf load_v16(const __bf16* p) {
    return *(const v16bf*)p;
}
static __device__ __forceinline__ v8bf load_v8(const __bf16* p) {
    return *(const v8bf*)p;
}
static __device__ __forceinline__ v16bf cat16(v8bf a, v8bf b) {
    return __builtin_shufflevector(a, b, 0,1,2,3,4,5,6,7,8,9,10,11,12,13,14,15);
}
static __device__ __forceinline__ v8f wmma_bf16(v16bf a, v16bf b, v8f c) {
    return __builtin_amdgcn_wmma_f32_16x16x32_bf16(false, a, false, b, (short)0, c, false, false);
}
static __device__ __forceinline__ int iclamp(int v, int lo, int hi) {
    return v < lo ? lo : (v > hi ? hi : v);
}
// Low 32 bits of a generic (shared-aperture) pointer == LDS byte offset.
static __device__ __forceinline__ unsigned lds_off(const void* p) {
    return (unsigned)(unsigned long long)p;
}
// Per-lane 16B async copy global -> LDS (tracked by ASYNCcnt).
static __device__ __forceinline__ void async_ld_b128(unsigned lds_byte, const void* gptr) {
    asm volatile("global_load_async_to_lds_b128 %0, %1, off"
                 :: "v"(lds_byte), "v"(gptr) : "memory");
}

// ---------------------------------------------------------------------------
// Kernel 0: fp32 -> bf16 conversion (grid-stride)
// ---------------------------------------------------------------------------
__global__ void cvt_f32_bf16(const float* __restrict__ in, __bf16* __restrict__ out, int n) {
    int i = blockIdx.x * blockDim.x + threadIdx.x;
    int stride = gridDim.x * blockDim.x;
    for (; i < n; i += stride) out[i] = (__bf16)in[i];
}

// ---------------------------------------------------------------------------
// Kernel 1: QKV projection GEMM. One wave computes a 16(M) x 64(N) strip
// (4 C tiles, A-fragment reused 4x -> 40B operand traffic per WMMA).
// The 64-wide N strip aligns exactly with one head.
//   Q,K stored [B,H,S,64] bf16 (Q pre-scaled by 1/8); V stored transposed
//   [B,H,64,S] bf16 so the attention PV B-fragments are contiguous.
// ---------------------------------------------------------------------------
__global__ void qkv_kernel(const __bf16* __restrict__ X,
                           const __bf16* __restrict__ Wq,
                           const __bf16* __restrict__ Wk,
                           const __bf16* __restrict__ Wv,
                           const float* __restrict__ bq,
                           const float* __restrict__ bk,
                           const float* __restrict__ bv,
                           __bf16* __restrict__ Qo,
                           __bf16* __restrict__ Ko,
                           __bf16* __restrict__ Vt) {
    const int lane = threadIdx.x & 31;
    const int wid  = threadIdx.x >> 5;
    const int task = blockIdx.x * 8 + wid;

    const int MT = (BATCH * S_LEN) / 16;   // 512
    const int NS = E_DIM / 64;             // 12 strips (== heads)
    const int TILES = MT * NS;             // 6144 per matrix

    const int which  = task / TILES;       // 0=Q 1=K 2=V
    const int t      = task % TILES;
    const int mtile  = t / NS;
    const int nstrip = t % NS;             // == head index

    const __bf16* Wm  = (which == 0) ? Wq : (which == 1) ? Wk : Wv;
    const float*  bia = (which == 0) ? bq : (which == 1) ? bk : bv;

    const int half = lane >> 4;
    const int l15  = lane & 15;

    const int m = mtile * 16 + l15;        // A row (flat seq index)
    const __bf16* arow = X + (long)m * E_DIM;
    const __bf16* brow0 = Wm + (long)(nstrip * 64 + l15) * E_DIM;

    v8f acc[4] = {{}, {}, {}, {}};
#pragma unroll 2
    for (int kb = 0; kb < E_DIM; kb += 32) {
        v16bf a = load_v16(arow + kb + half * 16);
#pragma unroll
        for (int sub = 0; sub < 4; sub++) {
            v16bf b = load_v16(brow0 + (long)sub * 16 * E_DIM + kb + half * 16);
            acc[sub] = wmma_bf16(a, b, acc[sub]);
        }
    }

    const float scale = (which == 0) ? 0.125f : 1.0f;   // 1/sqrt(64) for Q
    const int b_ = (mtile * 16) >> 12;                  // batch
    const int s0 = ((mtile * 16) & (S_LEN - 1)) + 8 * half;
    const int bh = b_ * H_NUM + nstrip;

#pragma unroll
    for (int sub = 0; sub < 4; sub++) {
        const int hd   = sub * 16 + l15;
        const float bias = bia[nstrip * 64 + hd];
        if (which < 2) {
            __bf16* dst = ((which == 0) ? Qo : Ko) +
                          (((long)bh * S_LEN + s0) * HD_DIM + hd);
#pragma unroll
            for (int r = 0; r < 8; r++)
                dst[(long)r * HD_DIM] = (__bf16)((acc[sub][r] + bias) * scale);
        } else {
            v8bf pack;
#pragma unroll
            for (int r = 0; r < 8; r++) pack[r] = (__bf16)(acc[sub][r] + bias);
            __bf16* dst = Vt + (((long)bh * HD_DIM + hd) * S_LEN + s0);
            *(v8bf*)dst = pack;
        }
    }
}

// ---------------------------------------------------------------------------
// Kernel 2: banded flash attention, one wave per (b,h,16-query tile).
// 17 iterations x 32 keys cover tiles [qt-17, qt+16] (superset of the +-256
// band). K tiles (32x64) and V tiles (64x32, from V^T) are double-buffered in
// LDS via GLOBAL_LOAD_ASYNC_TO_LDS_B128 (16 x 16B per tile pair per wave),
// staged one iteration ahead and consumed after s_wait_asynccnt. Fully-OOB
// 16-element groups are address-clamped; the band mask forces their P to 0.
// ---------------------------------------------------------------------------
__global__ void attn_kernel(const __bf16* __restrict__ Q,
                            const __bf16* __restrict__ K,
                            const __bf16* __restrict__ Vt,
                            float* __restrict__ out) {
    __shared__ __bf16 kbuf[8][2][32 * 64];   // 8 waves x double-buffered 4KB
    __shared__ __bf16 vbuf[8][2][64 * 32];   // 8 waves x double-buffered 4KB
    __shared__ __bf16 pbuf[8][16 * 32];      // per-wave P staging (1KB)

    const int lane = threadIdx.x & 31;
    const int wid  = threadIdx.x >> 5;
    const int task = blockIdx.x * 8 + wid;

    const int qt = task & 255;
    const int bh = task >> 8;
    const int b_ = bh / H_NUM;
    const int h_ = bh % H_NUM;

    const __bf16* Qp = Q  + (long)bh * S_LEN * HD_DIM;
    const __bf16* Kp = K  + (long)bh * S_LEN * HD_DIM;
    const __bf16* Vp = Vt + (long)bh * HD_DIM * S_LEN;

    const int half  = lane >> 4;
    const int l15   = lane & 15;
    const int qbase = qt * 16;

    const unsigned kldsB[2] = { lds_off(&kbuf[wid][0][0]), lds_off(&kbuf[wid][1][0]) };
    const unsigned vldsB[2] = { lds_off(&vbuf[wid][0][0]), lds_off(&vbuf[wid][1][0]) };

    // ---- async stage of one iteration's K/V tiles into LDS buffer `buf` ----
    auto stage = [&](int buf, int kb) {
        // K tile: 32 rows x 128B; unit u in [0,256): row=u>>3, 16B slot u&7.
#pragma unroll
        for (int i = 0; i < 8; i++) {
            const int u   = i * 32 + lane;
            const int row = iclamp(kb + (u >> 3), 0, S_LEN - 1);
            async_ld_b128(kldsB[buf] + u * 16,
                          (const char*)Kp + (long)row * 128 + (u & 7) * 16);
        }
        // V tile: 64 hd-rows x 64B; unit u in [0,256): row=u>>2, qtr=u&3.
        // Each 32B half-row (16 keys) clamps independently (16-aligned).
#pragma unroll
        for (int i = 0; i < 8; i++) {
            const int u   = i * 32 + lane;
            const int row = u >> 2, qtr = u & 3;
            const int ks  = iclamp(kb + (qtr >> 1) * 16, 0, S_LEN - 16);
            async_ld_b128(vldsB[buf] + u * 16,
                          (const char*)Vp + (long)row * (S_LEN * 2) + ks * 2 + (qtr & 1) * 16);
        }
    };

    // Q A-fragments (hd 0..31 and 32..63), kept in registers.
    const __bf16* qrow = Qp + (long)(qbase + l15) * HD_DIM;
    const v16bf qa0 = load_v16(qrow + half * 16);
    const v16bf qa1 = load_v16(qrow + 32 + half * 16);

    v8f o0 = {}, o1 = {}, o2 = {}, o3 = {};
    float mrun[8], lrun[8];
#pragma unroll
    for (int r = 0; r < 8; r++) { mrun[r] = -1e30f; lrun[r] = 0.0f; }

    const int Mb = qbase + 8 * half;       // global row of element r is Mb + r
    const int kb0 = qbase - 272;

    stage(0, kb0);                          // prime the pipeline

    for (int j = 0; j < 17; j++) {
        const int kb  = kb0 + 32 * j;
        const int buf = j & 1;

        if (j < 16) {
            stage(buf ^ 1, kb + 32);        // stage next iteration
            // warm L2 for iteration j+2's K rows
            __builtin_prefetch((const char*)Kp +
                               (long)iclamp(kb + 64 + lane, 0, S_LEN - 1) * 128, 0, 1);
            asm volatile("s_wait_asynccnt 0x10" ::: "memory");  // buffer j done
        } else {
            asm volatile("s_wait_asynccnt 0x0" ::: "memory");
        }

        const __bf16* kw = &kbuf[wid][buf][0];
        const __bf16* vw = &vbuf[wid][buf][0];

        // ---- scores: two 16x16 tiles (keys kb..kb+15, kb+16..kb+31) ----
        v8f s0 = {}, s1 = {};
        {
            const __bf16* kr = kw + (long)l15 * 64;
            s0 = wmma_bf16(qa0, load_v16(kr + half * 16), s0);
            s0 = wmma_bf16(qa1, load_v16(kr + 32 + half * 16), s0);
        }
        {
            const __bf16* kr = kw + (long)(16 + l15) * 64;
            s1 = wmma_bf16(qa0, load_v16(kr + half * 16), s1);
            s1 = wmma_bf16(qa1, load_v16(kr + 32 + half * 16), s1);
        }

        // ---- band + range mask ----
        const int n0 = kb + l15;
        const int n1 = kb + 16 + l15;
#pragma unroll
        for (int r = 0; r < 8; r++) {
            const int mg = Mb + r;
            const int d0 = n0 - mg, d1 = n1 - mg;
            const bool v0 = ((unsigned)n0 < (unsigned)S_LEN) && (d0 <= WIN) && (d0 >= -WIN);
            const bool v1 = ((unsigned)n1 < (unsigned)S_LEN) && (d1 <= WIN) && (d1 >= -WIN);
            s0[r] = v0 ? s0[r] : -1e30f;
            s1[r] = v1 ? s1[r] : -1e30f;
        }

        // ---- online softmax (row stats across 32 cols via xor-shuffles) ----
#pragma unroll
        for (int r = 0; r < 8; r++) {
            float mx = fmaxf(s0[r], s1[r]);
#pragma unroll
            for (int d = 1; d < 16; d <<= 1) mx = fmaxf(mx, __shfl_xor(mx, d, 32));
            const float mnew  = fmaxf(mrun[r], mx);
            const float alpha = __expf(mrun[r] - mnew);
            const float p0 = (s0[r] < -1e29f) ? 0.0f : __expf(s0[r] - mnew);
            const float p1 = (s1[r] < -1e29f) ? 0.0f : __expf(s1[r] - mnew);
            s0[r] = p0; s1[r] = p1;
            float rs = p0 + p1;
#pragma unroll
            for (int d = 1; d < 16; d <<= 1) rs += __shfl_xor(rs, d, 32);
            lrun[r] = lrun[r] * alpha + rs;
            mrun[r] = mnew;
            o0[r] *= alpha; o1[r] *= alpha; o2[r] *= alpha; o3[r] *= alpha;
        }

        // ---- P (C-layout) -> LDS -> A-layout fragment ----
#pragma unroll
        for (int r = 0; r < 8; r++) {
            pbuf[wid][(r + 8 * half) * 32 + l15]      = (__bf16)s0[r];
            pbuf[wid][(r + 8 * half) * 32 + 16 + l15] = (__bf16)s1[r];
        }
        __builtin_amdgcn_wave_barrier();
        const __bf16* pr = &pbuf[wid][l15 * 32];
        const v16bf pa = load_v16(pr + half * 16);

        // ---- PV: O(16x64) += P(16x32) x V(32x64), 4 hd chunks of 16 ----
        o0 = wmma_bf16(pa, load_v16(vw + (long)(l15)      * 32 + half * 16), o0);
        o1 = wmma_bf16(pa, load_v16(vw + (long)(16 + l15) * 32 + half * 16), o1);
        o2 = wmma_bf16(pa, load_v16(vw + (long)(32 + l15) * 32 + half * 16), o2);
        o3 = wmma_bf16(pa, load_v16(vw + (long)(48 + l15) * 32 + half * 16), o3);
    }

    // ---- epilogue: divide by row sums, store fp32 [B,S,E] ----
    const long ob = ((long)b_ * S_LEN + qbase + 8 * half) * E_DIM + h_ * HD_DIM;
#pragma unroll
    for (int r = 0; r < 8; r++) {
        const float inv = 1.0f / lrun[r];
        const long row = ob + (long)r * E_DIM;
        out[row + l15]      = o0[r] * inv;
        out[row + 16 + l15] = o1[r] * inv;
        out[row + 32 + l15] = o2[r] * inv;
        out[row + 48 + l15] = o3[r] * inv;
    }
}

// ---------------------------------------------------------------------------
extern "C" void kernel_launch(void* const* d_in, const int* in_sizes, int n_in,
                              void* d_out, int out_size, void* d_ws, size_t ws_size,
                              hipStream_t stream) {
    const float* x_f  = (const float*)d_in[0];   // [2,4096,768]
    const float* Wq_f = (const float*)d_in[1];   // [768,768]
    const float* Wk_f = (const float*)d_in[2];
    const float* Wv_f = (const float*)d_in[3];
    const float* bq   = (const float*)d_in[4];
    const float* bk   = (const float*)d_in[5];
    const float* bv   = (const float*)d_in[6];
    float* out = (float*)d_out;

    const long NX = (long)BATCH * S_LEN * E_DIM;           // 6,291,456
    const long NW = (long)E_DIM * E_DIM;                   // 589,824
    const long NQ = (long)BATCH * H_NUM * S_LEN * HD_DIM;  // 6,291,456

    char* w = (char*)d_ws;
    __bf16* Xb  = (__bf16*)w;                 w += NX * 2;
    __bf16* Wqb = (__bf16*)w;                 w += NW * 2;
    __bf16* Wkb = (__bf16*)w;                 w += NW * 2;
    __bf16* Wvb = (__bf16*)w;                 w += NW * 2;
    __bf16* Qb  = (__bf16*)w;                 w += NQ * 2;
    __bf16* Kb  = (__bf16*)w;                 w += NQ * 2;
    __bf16* Vtb = (__bf16*)w;                 w += NQ * 2;

    // 1) fp32 -> bf16
    cvt_f32_bf16<<<4096, 256, 0, stream>>>(x_f,  Xb,  (int)NX);
    cvt_f32_bf16<<<1024, 256, 0, stream>>>(Wq_f, Wqb, (int)NW);
    cvt_f32_bf16<<<1024, 256, 0, stream>>>(Wk_f, Wkb, (int)NW);
    cvt_f32_bf16<<<1024, 256, 0, stream>>>(Wv_f, Wvb, (int)NW);

    // 2) QKV projections: 3 * 512 * 12 strip-tasks / 8 waves = 2304 blocks
    qkv_kernel<<<2304, 256, 0, stream>>>(Xb, Wqb, Wkb, Wvb, bq, bk, bv,
                                         Qb, Kb, Vtb);

    // 3) banded attention: 24 * 256 wave-tasks / 8 = 768 blocks
    attn_kernel<<<768, 256, 0, stream>>>(Qb, Kb, Vtb, out);
}
